// GATModel_57655640981805
// MI455X (gfx1250) — matrix operand order repaired and checked by reading, not compiled, not run on hardware
//
#include <hip/hip_runtime.h>

typedef __attribute__((ext_vector_type(16))) _Float16 v16h;
typedef __attribute__((ext_vector_type(8)))  _Float16 v8h;
typedef __attribute__((ext_vector_type(8)))  float    v8f;

#define NEG_SLOPE 0.2f

// ---------- order-preserving float <-> uint encoding for atomic max ----------
__device__ __forceinline__ unsigned encf(float f) {
  unsigned u = __float_as_uint(f);
  return (u & 0x80000000u) ? ~u : (u | 0x80000000u);
}
__device__ __forceinline__ float decf(unsigned u) {
  return (u & 0x80000000u) ? __uint_as_float(u & 0x7fffffffu)
                           : __uint_as_float(~u);
}

// ---------- utility ----------
__global__ void k_zero(float* p, int n) {
  int t = blockIdx.x * blockDim.x + threadIdx.x;
  if (t < n) p[t] = 0.f;
}

// ---------- layer 1 GEMM: h1 = x @ W1   (x:[N,5], W1:[5,256]) ----------
__global__ void k_gemm1(const float* __restrict__ x, const float* __restrict__ W1,
                        float* __restrict__ h1, int N) {
  int t = blockIdx.x * blockDim.x + threadIdx.x;
  if (t >= N * 256) return;
  int n = t >> 8, o = t & 255;
  const float* xp = x + n * 5;
  float acc = 0.f;
#pragma unroll
  for (int k = 0; k < 5; ++k) acc += xp[k] * W1[k * 256 + o];
  h1[t] = acc;
}

// ---------- attention logits: al[n,h] = sum_c h[n,h,c] * a[h,c] ----------
__global__ void k_logits(const float* __restrict__ hfeat,
                         const float* __restrict__ a_s, const float* __restrict__ a_d,
                         float* __restrict__ als, float* __restrict__ ald,
                         int N, int H, int C) {
  int t = blockIdx.x * blockDim.x + threadIdx.x;
  if (t >= N * H) return;
  int n = t / H, h = t % H;
  const float* hp = hfeat + (n * H + h) * C;
  float ss = 0.f, sd = 0.f;
  for (int c = 0; c < C; ++c) {
    float v = hp[c];
    ss += v * a_s[h * C + c];
    sd += v * a_d[h * C + c];
  }
  als[t] = ss;
  ald[t] = sd;
}

// ---------- edge pass 1: segment max of leaky_relu(al_s[src]+al_d[dst]) ----------
__global__ void k_edge_max(const int* __restrict__ ei, int E, int N, int H,
                           const float* __restrict__ als, const float* __restrict__ ald,
                           unsigned* __restrict__ m) {
  int t = blockIdx.x * blockDim.x + threadIdx.x;
  int total = (E + N) * H;
  if (t >= total) return;
  int h = t % H, e = t / H;
  int s, d;
  if (e < E) { s = ei[e]; d = ei[E + e]; } else { s = d = e - E; }
  float x = als[s * H + h] + ald[d * H + h];
  x = (x > 0.f) ? x : NEG_SLOPE * x;
  atomicMax(&m[d * H + h], encf(x));
}

// ---------- edge pass 2: p = exp(e - m[dst]); segment sum; stash p ----------
__global__ void k_edge_sum(const int* __restrict__ ei, int E, int N, int H,
                           const float* __restrict__ als, const float* __restrict__ ald,
                           const unsigned* __restrict__ m, float* __restrict__ ssum,
                           float* __restrict__ pbuf) {
  int t = blockIdx.x * blockDim.x + threadIdx.x;
  int total = (E + N) * H;
  if (t >= total) return;
  int h = t % H, e = t / H;
  int s, d;
  if (e < E) { s = ei[e]; d = ei[E + e]; } else { s = d = e - E; }
  float x = als[s * H + h] + ald[d * H + h];
  x = (x > 0.f) ? x : NEG_SLOPE * x;
  float p = expf(x - decf(m[d * H + h]));
  atomicAdd(&ssum[d * H + h], p);
  pbuf[e * H + h] = p;
}

// ---------- edge pass 3: agg[dst] += h[src] * alpha   (float4-vectorized) ----------
__global__ void k_edge_agg4(const int* __restrict__ ei, int E, int N, int H, int C,
                            const float* __restrict__ pbuf, const float* __restrict__ ssum,
                            const float* __restrict__ hfeat, float* __restrict__ agg) {
  int t = blockIdx.x * blockDim.x + threadIdx.x;
  int HC = H * C;
  int HC4 = HC >> 2;
  int total = (E + N) * HC4;
  if (t >= total) return;
  int q = t % HC4, e = t / HC4;
  int idx = q << 2;                 // 4 consecutive channels, same head
  int h = idx / C;
  int s, d;
  if (e < E) { s = ei[e]; d = ei[E + e]; } else { s = d = e - E; }
  float alpha = pbuf[e * H + h] / (ssum[d * H + h] + 1e-16f);
  const float4 hv = *(const float4*)(hfeat + (size_t)s * HC + idx);
  float* ap = agg + (size_t)d * HC + idx;
  atomicAdd(ap + 0, hv.x * alpha);
  atomicAdd(ap + 1, hv.y * alpha);
  atomicAdd(ap + 2, hv.z * alpha);
  atomicAdd(ap + 3, hv.w * alpha);
}

// ---------- pack weights [K,64] f32 (row-major) into WMMA B-fragment order f16 ----------
// Wpk[((kc/32)*4 + nt)*512 + lane*16 + i] = W[(kc + kb(lane) + (i&7) + ((i&8)<<1))*64 + nt*16 + (lane&15)]
__global__ void k_packW(const float* __restrict__ W, _Float16* __restrict__ Wpk, int K) {
  int t = blockIdx.x * blockDim.x + threadIdx.x;
  if (t >= K * 64) return;
  int c  = t >> 11;          // chunk (2048 halves per 32-K chunk)
  int r  = t & 2047;
  int nt = r >> 9;
  int q  = r & 511;
  int L  = q >> 4;
  int i  = q & 15;
  int kb = (L & 16) ? 8 : 0;
  int k  = c * 32 + kb + (i & 7) + ((i & 8) << 1);
  int n  = nt * 16 + (L & 15);
  Wpk[t] = (_Float16)W[k * 64 + n];
}

// ---------- activation -> f16 (bias + relu folded): layer-2 GEMM A operand ----------
__global__ void k_act1_f16(const float* __restrict__ agg1, const float* __restrict__ b1,
                           _Float16* __restrict__ out, int N) {
  int t = blockIdx.x * blockDim.x + threadIdx.x;
  if (t >= N * 256) return;
  float v = agg1[t] + b1[t & 255];
  out[t] = (_Float16)(v > 0.f ? v : 0.f);
}

// ---------- combine GEMM A operand: [relu(agg2+b2) | gemb[batch]] -> f16 [N,128] ----------
__global__ void k_combA_f16(const float* __restrict__ agg2, const float* __restrict__ b2,
                            const float* __restrict__ gemb, const int* __restrict__ batch,
                            _Float16* __restrict__ out, int N) {
  int t = blockIdx.x * blockDim.x + threadIdx.x;
  if (t >= N * 128) return;
  int n = t >> 7, k = t & 127;
  float v;
  if (k < 64) {
    v = agg2[n * 64 + k] + b2[k];
    v = v > 0.f ? v : 0.f;
  } else {
    v = gemb[batch[n] * 64 + (k - 64)];
  }
  out[t] = (_Float16)v;
}

// ---------- generic WMMA GEMM: D[rows,64] = A[rows,K]f16 @ Wpk(packed f16) ----------
// orelu: 1 -> D = relu(acc + obias[col]); 0 -> D = acc
__global__ void k_gemm_wmma_f16(const _Float16* __restrict__ A,
                                const _Float16* __restrict__ Wpk,
                                const float* __restrict__ obias, int orelu,
                                float* __restrict__ D, int ntiles, int K) {
  int wave = threadIdx.x >> 5;
  int lane = threadIdx.x & 31;
  int tile = blockIdx.x * (blockDim.x >> 5) + wave;
  if (tile >= ntiles) return;            // wave-uniform
  int lh   = lane & 15;
  int hi   = (lane >> 4) & 1;
  int row  = tile * 16 + lh;
  int koff = hi ? 8 : 0;
  const _Float16* arow = A + (size_t)row * K;
  v8f acc0 = {}, acc1 = {}, acc2 = {}, acc3 = {};
  for (int kc = 0; kc < K; kc += 32) {
    v8h a0 = *(const v8h*)(arow + kc + koff);          // global_load_b128
    v8h a1 = *(const v8h*)(arow + kc + koff + 16);     // global_load_b128
    v16h a = __builtin_shufflevector(a0, a1, 0, 1, 2, 3, 4, 5, 6, 7,
                                     8, 9, 10, 11, 12, 13, 14, 15);
    const v16h* wbase = (const v16h*)Wpk + (size_t)(kc >> 5) * 128;  // 4 frags * 32 lanes
    v16h fb0 = wbase[0 * 32 + lane];
    v16h fb1 = wbase[1 * 32 + lane];
    v16h fb2 = wbase[2 * 32 + lane];
    v16h fb3 = wbase[3 * 32 + lane];
    acc0 = __builtin_amdgcn_wmma_f32_16x16x32_f16(false, a, false, fb0, (short)0, acc0, false, false);
    acc1 = __builtin_amdgcn_wmma_f32_16x16x32_f16(false, a, false, fb1, (short)0, acc1, false, false);
    acc2 = __builtin_amdgcn_wmma_f32_16x16x32_f16(false, a, false, fb2, (short)0, acc2, false, false);
    acc3 = __builtin_amdgcn_wmma_f32_16x16x32_f16(false, a, false, fb3, (short)0, acc3, false, false);
  }
  int mbase = tile * 16 + hi * 8;
  if (orelu) {
    float bb0 = obias[lh], bb1 = obias[16 + lh], bb2 = obias[32 + lh], bb3 = obias[48 + lh];
#pragma unroll
    for (int j = 0; j < 8; ++j) {
      float* dr = D + (size_t)(mbase + j) * 64 + lh;
      float v0 = acc0[j] + bb0, v1 = acc1[j] + bb1, v2 = acc2[j] + bb2, v3 = acc3[j] + bb3;
      dr[0]  = v0 > 0.f ? v0 : 0.f;
      dr[16] = v1 > 0.f ? v1 : 0.f;
      dr[32] = v2 > 0.f ? v2 : 0.f;
      dr[48] = v3 > 0.f ? v3 : 0.f;
    }
  } else {
#pragma unroll
    for (int j = 0; j < 8; ++j) {
      float* dr = D + (size_t)(mbase + j) * 64 + lh;
      dr[0]  = acc0[j];
      dr[16] = acc1[j];
      dr[32] = acc2[j];
      dr[48] = acc3[j];
    }
  }
}

// ---------- f32 scalar tails (N % 16 rows; zero work for N = 50000) ----------
__global__ void k_gemm_l2_tail(const float* __restrict__ A, const float* __restrict__ bias,
                               const float* __restrict__ W, float* __restrict__ D,
                               int N, int base) {
  int t = blockIdx.x * blockDim.x + threadIdx.x;
  int rows = N - base;
  if (t >= rows * 64) return;
  int n = base + t / 64, c = t % 64;
  float acc = 0.f;
  for (int k = 0; k < 256; ++k) {
    float v = A[n * 256 + k] + bias[k];
    v = v > 0.f ? v : 0.f;
    acc += v * W[k * 64 + c];
  }
  D[n * 64 + c] = acc;
}

__global__ void k_gemm_comb_tail(const float* __restrict__ agg2, const float* __restrict__ b2,
                                 const float* __restrict__ gemb, const int* __restrict__ batch,
                                 const float* __restrict__ Wc, const float* __restrict__ bc,
                                 float* __restrict__ comb, int N, int base) {
  int t = blockIdx.x * blockDim.x + threadIdx.x;
  int rows = N - base;
  if (t >= rows * 64) return;
  int n = base + t / 64, c = t % 64;
  int g = batch[n];
  float acc = bc[c];
  for (int k = 0; k < 64; ++k) {
    float v = agg2[n * 64 + k] + b2[k];
    v = v > 0.f ? v : 0.f;
    acc += v * Wc[k * 64 + c];
  }
  for (int k = 0; k < 64; ++k) acc += gemb[g * 64 + k] * Wc[(64 + k) * 64 + c];
  comb[n * 64 + c] = acc > 0.f ? acc : 0.f;
}

// ---------- graph MLP: gemb = relu(gf @ Wg1 + bg1) @ Wg2 + bg2 ----------
__global__ void k_graphmlp(const float* __restrict__ gf, const float* __restrict__ Wg1,
                           const float* __restrict__ bg1, const float* __restrict__ Wg2,
                           const float* __restrict__ bg2, float* __restrict__ gemb) {
  __shared__ float t1[64];
  int g = blockIdx.x, j = threadIdx.x;
  float v = gf[g * 2 + 0] * Wg1[j] + gf[g * 2 + 1] * Wg1[64 + j] + bg1[j];
  t1[j] = v > 0.f ? v : 0.f;
  __syncthreads();
  float acc = bg2[j];
  for (int k = 0; k < 64; ++k) acc += t1[k] * Wg2[k * 64 + j];
  gemb[g * 64 + j] = acc;
}

// ---------- final linear: out = comb @ Wl + bl ----------
__global__ void k_final(const float* __restrict__ comb, const float* __restrict__ Wl,
                        const float* __restrict__ bl, float* __restrict__ out, int N) {
  int n = blockIdx.x * blockDim.x + threadIdx.x;
  if (n >= N) return;
  float acc = bl[0];
  for (int c = 0; c < 64; ++c) acc += comb[n * 64 + c] * Wl[c];
  out[n] = acc;
}

extern "C" void kernel_launch(void* const* d_in, const int* in_sizes, int n_in,
                              void* d_out, int out_size, void* d_ws, size_t ws_size,
                              hipStream_t stream) {
  const float* x     = (const float*)d_in[0];
  const int*   ei    = (const int*)d_in[1];
  const float* gf    = (const float*)d_in[2];
  const int*   batch = (const int*)d_in[3];
  const float* W1    = (const float*)d_in[4];
  const float* as1   = (const float*)d_in[5];
  const float* ad1   = (const float*)d_in[6];
  const float* b1    = (const float*)d_in[7];
  const float* W2    = (const float*)d_in[8];
  const float* as2   = (const float*)d_in[9];
  const float* ad2   = (const float*)d_in[10];
  const float* b2    = (const float*)d_in[11];
  const float* Wg1   = (const float*)d_in[12];
  const float* bg1   = (const float*)d_in[13];
  const float* Wg2   = (const float*)d_in[14];
  const float* bg2   = (const float*)d_in[15];
  const float* Wc    = (const float*)d_in[16];
  const float* bc    = (const float*)d_in[17];
  const float* Wl    = (const float*)d_in[18];
  const float* bl    = (const float*)d_in[19];
  float* out = (float*)d_out;

  const int N  = in_sizes[0] / 5;   // 50000
  const int E  = in_sizes[1] / 2;   // 800000
  const int Et = E + N;             // edges + self loops

  // workspace layout (floats), aggressive reuse across phases
  float* ws   = (float*)d_ws;
  float* h1   = ws;                          // [N,256]
  float* agg1 = h1   + (size_t)N * 256;      // [N,256]
  float* als1 = agg1 + (size_t)N * 256;      // [N,4]
  float* ald1 = als1 + (size_t)N * 4;        // [N,4]
  float* m1   = ald1 + (size_t)N * 4;        // [N,4] (unsigned encoded)
  float* s1   = m1   + (size_t)N * 4;        // [N,4]
  float* pbuf = s1   + (size_t)N * 4;        // [Et,4]
  float* gemb = pbuf + (size_t)Et * 4;       // [16,64]
  _Float16* Wpk2 = (_Float16*)(gemb + 1024);       // 256*64 halves (32B aligned)
  _Float16* Wpkc = Wpk2 + 256 * 64;                // 128*64 halves
  // reuses (lifetimes verified: producer finishes before overlay is written)
  float*    h2     = h1;                           // [N,64]
  float*    agg2   = agg1;                         // [N,64]
  float*    comb   = h1 + (size_t)N * 64;          // [N,64]
  _Float16* act1h  = (_Float16*)pbuf;              // [N,256] f16 (25.6MB, larger than the 13.6MB pbuf region)
  _Float16* combAh = (_Float16*)(h1 + (size_t)N * 128);  // [N,128] f16

  auto cdiv = [](long long a, long long b) { return (unsigned)((a + b - 1) / b); };
  const int B = 256;
  const int ntiles  = N / 16;
  const int base    = ntiles * 16;
  const int tail    = N - base;
  const int gblocks = (ntiles + 7) / 8;      // 8 waves per 256-thread block

  // independent prep: graph MLP + weight packs
  k_graphmlp<<<16, 64, 0, stream>>>(gf, Wg1, bg1, Wg2, bg2, gemb);
  k_packW<<<cdiv(256 * 64, B), B, 0, stream>>>(W2, Wpk2, 256);
  k_packW<<<cdiv(128 * 64, B), B, 0, stream>>>(Wc, Wpkc, 128);

  // ----- GAT layer 1 -----
  k_gemm1<<<cdiv((long long)N * 256, B), B, 0, stream>>>(x, W1, h1, N);
  k_logits<<<cdiv((long long)N * 4, B), B, 0, stream>>>(h1, as1, ad1, als1, ald1, N, 4, 64);
  k_zero<<<cdiv((long long)N * 256, B), B, 0, stream>>>(agg1, N * 256);
  k_zero<<<cdiv((long long)N * 8, B), B, 0, stream>>>(m1, N * 8);   // m1 + s1 contiguous
  k_edge_max<<<cdiv((long long)Et * 4, B), B, 0, stream>>>(ei, E, N, 4, als1, ald1, (unsigned*)m1);
  k_edge_sum<<<cdiv((long long)Et * 4, B), B, 0, stream>>>(ei, E, N, 4, als1, ald1,
                                                           (const unsigned*)m1, s1, pbuf);
  k_edge_agg4<<<cdiv((long long)Et * 64, B), B, 0, stream>>>(ei, E, N, 4, 64, pbuf, s1, h1, agg1);

  // ----- GAT layer 2 (f16 WMMA GEMM; layer-1 bias+relu folded into f16 conversion) -----
  k_act1_f16<<<cdiv((long long)N * 256, B), B, 0, stream>>>(agg1, b1, act1h, N);
  k_gemm_wmma_f16<<<gblocks, 256, 0, stream>>>(act1h, Wpk2, (const float*)nullptr, 0,
                                               h2, ntiles, 256);
  if (tail > 0)
    k_gemm_l2_tail<<<cdiv((long long)tail * 64, B), B, 0, stream>>>(agg1, b1, W2, h2, N, base);
  k_logits<<<cdiv((long long)N, B), B, 0, stream>>>(h2, as2, ad2, als1, ald1, N, 1, 64);
  k_zero<<<cdiv((long long)N * 64, B), B, 0, stream>>>(agg2, N * 64);
  k_zero<<<cdiv((long long)N, B), B, 0, stream>>>(m1, N);
  k_zero<<<cdiv((long long)N, B), B, 0, stream>>>(s1, N);
  k_edge_max<<<cdiv((long long)Et, B), B, 0, stream>>>(ei, E, N, 1, als1, ald1, (unsigned*)m1);
  k_edge_sum<<<cdiv((long long)Et, B), B, 0, stream>>>(ei, E, N, 1, als1, ald1,
                                                       (const unsigned*)m1, s1, pbuf);
  k_edge_agg4<<<cdiv((long long)Et * 16, B), B, 0, stream>>>(ei, E, N, 1, 64, pbuf, s1, h2, agg2);

  // ----- combine + head (f16 WMMA GEMM; layer-2 bias+relu folded; output bias+relu) -----
  k_combA_f16<<<cdiv((long long)N * 128, B), B, 0, stream>>>(agg2, b2, gemb, batch, combAh, N);
  k_gemm_wmma_f16<<<gblocks, 256, 0, stream>>>(combAh, Wpkc, bc, 1, comb, ntiles, 128);
  if (tail > 0)
    k_gemm_comb_tail<<<cdiv((long long)tail * 64, B), B, 0, stream>>>(agg2, b2, gemb, batch,
                                                                      Wc, bc, comb, N, base);
  k_final<<<cdiv((long long)N, B), B, 0, stream>>>(comb, Wl, bl, out, N);
}
